// SSM_82171314307104
// MI455X (gfx1250) — matrix operand order repaired and checked by reading, not compiled, not run on hardware
//
#include <hip/hip_runtime.h>
#include <hip/hip_bf16.h>

#define D_MODELC 1024
#define D_HIDDENC 1024
#define BATCHC 4
#define SEQC 4096
#define M_TOTAL (BATCHC * SEQC)     // 16384 rows (b,t)
#define CHUNK 128
#define NCHUNK (SEQC / CHUNK)       // 32

typedef __attribute__((ext_vector_type(16))) __bf16 v16bf;
typedef __attribute__((ext_vector_type(8)))  __bf16 v8bf;
typedef __attribute__((ext_vector_type(8)))  float  v8f;

#define WMMA_BF16(A_, B_, C_) \
  __builtin_amdgcn_wmma_f32_16x16x32_bf16(false, (A_), false, (B_), (short)0, (C_), false, false)

// ---------------------------------------------------------------- converts
__global__ void cvt_f32_bf16(const float* __restrict__ in, __bf16* __restrict__ out, int n) {
  for (int i = blockIdx.x * blockDim.x + threadIdx.x; i < n; i += gridDim.x * blockDim.x)
    out[i] = (__bf16)in[i];
}

// ------------------------------------------------- WMMA fragment loaders
// A fragment: 16x32 (MxK) bf16. Lane holds row M = lane&15,
// K = k0 + (lane>>4)*16 .. +16 contiguous halves (matches B fragment).
__device__ __forceinline__ v16bf load_a_frag(const __bf16* __restrict__ base, int lda,
                                             int m0, int k0, int lane) {
  const int row = m0 + (lane & 15);
  const int kh  = (lane >> 4) * 16;
  const __bf16* p = base + (size_t)row * lda + k0 + kh;
  v8bf lo = *(const v8bf*)(p);
  v8bf hi = *(const v8bf*)(p + 8);
  return __builtin_shufflevector(lo, hi, 0,1,2,3,4,5,6,7,8,9,10,11,12,13,14,15);
}

// B fragment: 32x16 (KxN) from a row-major [N,K] weight. Contiguous k order, same as A:
// lane L: N = L&15, K = k0 + (L>>4)*16 .. +16 contiguous halves.
__device__ __forceinline__ v16bf load_b_frag(const __bf16* __restrict__ base, int ldb,
                                             int n0, int k0, int lane) {
  const int row = n0 + (lane & 15);
  const int kh  = (lane >> 4) * 16;
  const v8bf* p = (const v8bf*)(base + (size_t)row * ldb + k0 + kh);
  v8bf lo = p[0];
  v8bf hi = p[1];
  return __builtin_shufflevector(lo, hi, 0,1,2,3,4,5,6,7,8,9,10,11,12,13,14,15);
}

// ---------------------------------------------------------- GEMM #1 (input projections)
// A_bar = exp(-softplus(x@dWt + db) * A) and bx = x@BWt + Bb, both [M,H] f32.
// Block = 256 thr = 8 waves in 2(M)x4(N); wave tile 64x32; block tile 128x128.
__global__ __launch_bounds__(256, 1)
void gemm_in_kernel(const __bf16* __restrict__ xb,       // [M, K] bf16
                    const __bf16* __restrict__ w_delta,  // [H, K] bf16
                    const __bf16* __restrict__ w_Bm,     // [H, K] bf16
                    const float* __restrict__ delta_b,
                    const float* __restrict__ B_b,
                    const float* __restrict__ Avec,
                    float* __restrict__ A_bar,           // [M, H]
                    float* __restrict__ bx)              // [M, H]
{
  const int K = D_MODELC, H = D_HIDDENC;
  const int lane  = threadIdx.x & 31;
  const int wave  = threadIdx.x >> 5;
  const int mBase = blockIdx.y * 128 + (wave & 1) * 64;
  const int nBase = blockIdx.x * 128 + (wave >> 1) * 32;

  v8f accD[4][2] = {{{0.f}}};
  v8f accB[4][2] = {{{0.f}}};

  for (int k0 = 0; k0 < K; k0 += 32) {
    v16bf a[4];
#pragma unroll
    for (int mt = 0; mt < 4; ++mt)
      a[mt] = load_a_frag(xb, K, mBase + mt * 16, k0, lane);
    v16bf d0 = load_b_frag(w_delta, K, nBase,      k0, lane);
    v16bf d1 = load_b_frag(w_delta, K, nBase + 16, k0, lane);
    v16bf b0 = load_b_frag(w_Bm,    K, nBase,      k0, lane);
    v16bf b1 = load_b_frag(w_Bm,    K, nBase + 16, k0, lane);

#pragma unroll
    for (int mt = 0; mt < 4; ++mt) {
      accD[mt][0] = WMMA_BF16(a[mt], d0, accD[mt][0]);
      accD[mt][1] = WMMA_BF16(a[mt], d1, accD[mt][1]);
      accB[mt][0] = WMMA_BF16(a[mt], b0, accB[mt][0]);
      accB[mt][1] = WMMA_BF16(a[mt], b1, accB[mt][1]);
    }
  }

  // Epilogue. C/D layout: lane L -> col N = L&15, rows M = (L>>4)*8 + v.
  const int col   = lane & 15;
  const int rhalf = (lane >> 4) * 8;
#pragma unroll
  for (int nt = 0; nt < 2; ++nt) {
    const int n  = nBase + nt * 16 + col;
    const float db = delta_b[n];
    const float bb = B_b[n];
    const float av = Avec[n];
#pragma unroll
    for (int mt = 0; mt < 4; ++mt) {
#pragma unroll
      for (int v = 0; v < 8; ++v) {
        const int m = mBase + mt * 16 + rhalf + v;
        const float z  = accD[mt][nt][v] + db;
        const float sp = (z > 20.0f) ? z : log1pf(__expf(z));
        A_bar[(size_t)m * H + n] = __expf(-sp * av);
        bx[(size_t)m * H + n]    = accB[mt][nt][v] + bb;
      }
    }
  }
}

// ---------------------------------------------------------------- chunked scan
__global__ void scan_pass1(const float* __restrict__ A_bar, const float* __restrict__ bx,
                           float* __restrict__ cp, float* __restrict__ cs) {
  const int tid = blockIdx.x * blockDim.x + threadIdx.x;   // (b*NCHUNK + c)*H + h
  if (tid >= BATCHC * NCHUNK * D_HIDDENC) return;
  const int h  = tid % D_HIDDENC;
  const int bc = tid / D_HIDDENC;
  const int c  = bc % NCHUNK;
  const int b  = bc / NCHUNK;
  size_t base = ((size_t)b * SEQC + (size_t)c * CHUNK) * D_HIDDENC + h;
  float p = 1.0f, s = 0.0f;
  for (int i = 0; i < CHUNK; ++i) {
    const float a = A_bar[base];
    const float x = bx[base];
    p *= a;
    s = a * s + x;
    base += D_HIDDENC;
  }
  cp[tid] = p;
  cs[tid] = s;
}

__global__ void scan_pass2(const float* __restrict__ cp, const float* __restrict__ cs,
                           float* __restrict__ cinit) {
  const int tid = blockIdx.x * blockDim.x + threadIdx.x;   // b*H + h
  if (tid >= BATCHC * D_HIDDENC) return;
  const int h = tid % D_HIDDENC;
  const int b = tid / D_HIDDENC;
  float st = 0.0f;
  for (int c = 0; c < NCHUNK; ++c) {
    const int idx = (b * NCHUNK + c) * D_HIDDENC + h;
    cinit[idx] = st;
    st = cp[idx] * st + cs[idx];
  }
}

__global__ void scan_pass3(const float* __restrict__ A_bar, const float* __restrict__ bx,
                           const float* __restrict__ cinit, __bf16* __restrict__ hout) {
  const int tid = blockIdx.x * blockDim.x + threadIdx.x;
  if (tid >= BATCHC * NCHUNK * D_HIDDENC) return;
  const int h  = tid % D_HIDDENC;
  const int bc = tid / D_HIDDENC;
  const int c  = bc % NCHUNK;
  const int b  = bc / NCHUNK;
  size_t base = ((size_t)b * SEQC + (size_t)c * CHUNK) * D_HIDDENC + h;
  float s = cinit[tid];
  for (int i = 0; i < CHUNK; ++i) {
    const float a = A_bar[base];
    const float x = bx[base];
    s = a * s + x;
    hout[base] = (__bf16)s;
    base += D_HIDDENC;
  }
}

// ---------------------------------------------------------- GEMM #2 (output)
// out[m,n] = sum_h h[m,h]*Cw[n,h] + sum_k x[m,k]*Dw[n,k] + Cb[n] + Db[n]
// Both GEMMs accumulate into the same f32 WMMA accumulator.
__global__ __launch_bounds__(256, 1)
void gemm_out_kernel(const __bf16* __restrict__ hb,   // [M, H]
                     const __bf16* __restrict__ xb,   // [M, K]
                     const __bf16* __restrict__ w_C,  // [D, H]
                     const __bf16* __restrict__ w_D,  // [D, K] (H==D here)
                     const float* __restrict__ C_b,
                     const float* __restrict__ D_b,
                     float* __restrict__ out)         // [M, D]
{
  const int K = D_MODELC;
  const int lane  = threadIdx.x & 31;
  const int wave  = threadIdx.x >> 5;
  const int mBase = blockIdx.y * 128 + (wave & 1) * 64;
  const int nBase = blockIdx.x * 128 + (wave >> 1) * 32;

  v8f acc[4][2] = {{{0.f}}};

  for (int k0 = 0; k0 < K; k0 += 32) {
    v16bf ah[4], ax[4];
#pragma unroll
    for (int mt = 0; mt < 4; ++mt) {
      ah[mt] = load_a_frag(hb, K, mBase + mt * 16, k0, lane);
      ax[mt] = load_a_frag(xb, K, mBase + mt * 16, k0, lane);
    }
    v16bf bc0 = load_b_frag(w_C, K, nBase,      k0, lane);
    v16bf bc1 = load_b_frag(w_C, K, nBase + 16, k0, lane);
    v16bf bd0 = load_b_frag(w_D, K, nBase,      k0, lane);
    v16bf bd1 = load_b_frag(w_D, K, nBase + 16, k0, lane);

#pragma unroll
    for (int mt = 0; mt < 4; ++mt) {
      acc[mt][0] = WMMA_BF16(ah[mt], bc0, acc[mt][0]);
      acc[mt][1] = WMMA_BF16(ah[mt], bc1, acc[mt][1]);
      acc[mt][0] = WMMA_BF16(ax[mt], bd0, acc[mt][0]);
      acc[mt][1] = WMMA_BF16(ax[mt], bd1, acc[mt][1]);
    }
  }

  const int col   = lane & 15;
  const int rhalf = (lane >> 4) * 8;
#pragma unroll
  for (int nt = 0; nt < 2; ++nt) {
    const int n = nBase + nt * 16 + col;
    const float bias = C_b[n] + D_b[n];
#pragma unroll
    for (int mt = 0; mt < 4; ++mt) {
#pragma unroll
      for (int v = 0; v < 8; ++v) {
        const int m = mBase + mt * 16 + rhalf + v;
        out[(size_t)m * D_MODELC + n] = acc[mt][nt][v] + bias;
      }
    }
  }
}

// ---------------------------------------------------------------- launcher
extern "C" void kernel_launch(void* const* d_in, const int* in_sizes, int n_in,
                              void* d_out, int out_size, void* d_ws, size_t ws_size,
                              hipStream_t stream) {
  (void)in_sizes; (void)n_in; (void)out_size; (void)ws_size;

  const float* x       = (const float*)d_in[0];
  const float* A       = (const float*)d_in[1];
  const float* B_w     = (const float*)d_in[2];
  const float* B_b     = (const float*)d_in[3];
  const float* C_w     = (const float*)d_in[4];
  const float* C_b     = (const float*)d_in[5];
  const float* D_w     = (const float*)d_in[6];
  const float* D_b     = (const float*)d_in[7];
  const float* delta_w = (const float*)d_in[8];
  const float* delta_b = (const float*)d_in[9];
  float* out = (float*)d_out;

  const size_t M = M_TOTAL, K = D_MODELC, H = D_HIDDENC;

  char* ws = (char*)d_ws;
  size_t off = 0;
  auto alloc = [&](size_t bytes) -> void* {
    void* p = ws + off;
    off = (off + bytes + 255) & ~(size_t)255;
    return p;
  };
  __bf16* xb    = (__bf16*)alloc(M * K * 2);            // 32 MB
  __bf16* wdel  = (__bf16*)alloc(H * K * 2);            // 2 MB
  __bf16* wB    = (__bf16*)alloc(H * K * 2);
  __bf16* wC    = (__bf16*)alloc(K * H * 2);
  __bf16* wD    = (__bf16*)alloc(H * K * 2);
  float*  A_bar = (float*) alloc(M * H * 4);            // 64 MB
  float*  bx    = (float*) alloc(M * H * 4);            // 64 MB
  __bf16* hb    = (__bf16*)alloc(M * H * 2);            // 32 MB
  float*  cp    = (float*) alloc((size_t)BATCHC * NCHUNK * H * 4);
  float*  cs    = (float*) alloc((size_t)BATCHC * NCHUNK * H * 4);
  float*  cini  = (float*) alloc((size_t)BATCHC * NCHUNK * H * 4);

  // 1) convert inputs to bf16
  cvt_f32_bf16<<<4096, 256, 0, stream>>>(x,       xb,   (int)(M * K));
  cvt_f32_bf16<<<1024, 256, 0, stream>>>(delta_w, wdel, (int)(H * K));
  cvt_f32_bf16<<<1024, 256, 0, stream>>>(B_w,     wB,   (int)(H * K));
  cvt_f32_bf16<<<1024, 256, 0, stream>>>(C_w,     wC,   (int)(K * H));
  cvt_f32_bf16<<<1024, 256, 0, stream>>>(D_w,     wD,   (int)(H * K));

  // 2) fused input projections (delta -> A_bar, B -> bx)
  dim3 gIn(H / 128, M / 128, 1);
  gemm_in_kernel<<<gIn, 256, 0, stream>>>(xb, wdel, wB, delta_b, B_b, A, A_bar, bx);

  // 3) chunked scan over time
  const int nScanThr = BATCHC * NCHUNK * D_HIDDENC;     // 131072
  scan_pass1<<<nScanThr / 256, 256, 0, stream>>>(A_bar, bx, cp, cs);
  scan_pass2<<<(BATCHC * D_HIDDENC) / 256, 256, 0, stream>>>(cp, cs, cini);
  scan_pass3<<<nScanThr / 256, 256, 0, stream>>>(A_bar, bx, cini, hb);

  // 4) fused output GEMMs: h@C^T + x@D^T + biases
  dim3 gOut(D_MODELC / 128, M / 128, 1);
  gemm_out_kernel<<<gOut, 256, 0, stream>>>(hb, xb, wC, wD, C_b, D_b, out);
}
